// DecomposeModel_35167192220017
// MI455X (gfx1250) — matrix operand order repaired and checked by reading, not compiled, not run on hardware
//
#include <hip/hip_runtime.h>
#include <hip/hip_bf16.h>

typedef _Float16 half_t;
typedef __attribute__((ext_vector_type(16))) _Float16 v16h;
typedef __attribute__((ext_vector_type(8)))  _Float16 v8h;
typedef __attribute__((ext_vector_type(8)))  float    v8f;

#define B_ROWS 8192
#define HDIM   1024

// ---------------------------------------------------------------------------
// CDNA5 async global->LDS copy (ASYNCcnt path), per CDNA5 ISA.
// ---------------------------------------------------------------------------
__device__ __forceinline__ void async_ld_b128(unsigned lds_off, const half_t* g) {
  asm volatile("global_load_async_to_lds_b128 %0, %1, off"
               :: "v"(lds_off), "v"(g) : "memory");
}
__device__ __forceinline__ void wait_async0() {
  asm volatile("s_wait_asynccnt 0x0" ::: "memory");
}
__device__ __forceinline__ unsigned lds_off_u32(const void* p) {
  // Flat LDS addresses map to LDS by discarding the upper 32 bits (ISA 10.2).
  return (unsigned)(unsigned long long)p;
}

// ---------------------------------------------------------------------------
// Pack kernels: f32 -> f16 workspace buffers
// ---------------------------------------------------------------------------
__global__ void k_pack_x1(const float* __restrict__ hid,
                          const float* __restrict__ ctx,
                          const float* __restrict__ lab,
                          half_t* __restrict__ x1) {
  size_t idx = (size_t)blockIdx.x * blockDim.x + threadIdx.x;
  size_t total = (size_t)B_ROWS * 3 * HDIM;
  if (idx >= total) return;
  int row = (int)(idx / (3 * HDIM));
  int col = (int)(idx % (3 * HDIM));
  float v;
  if (col < HDIM)            v = hid[(size_t)row * HDIM + col];
  else if (col < 2 * HDIM)   v = ctx[(size_t)row * HDIM + (col - HDIM)];
  else                       v = lab[(size_t)row * HDIM + (col - 2 * HDIM)];
  x1[idx] = (half_t)v;
}

// left_embedding -> columns [1024, 2048) of X2r (ld = 2048)
__global__ void k_pack_left(const float* __restrict__ left,
                            half_t* __restrict__ x2r) {
  size_t idx = (size_t)blockIdx.x * blockDim.x + threadIdx.x;
  if (idx >= (size_t)B_ROWS * HDIM) return;
  int row = (int)(idx / HDIM);
  int col = (int)(idx % HDIM);
  x2r[(size_t)row * (2 * HDIM) + HDIM + col] = (half_t)left[idx];
}

// ---------------------------------------------------------------------------
// Tiled transpose + convert: src f32 (K x N, row-major) -> dst f16 (N x K)
// ---------------------------------------------------------------------------
__global__ void k_transpose_f32_f16(const float* __restrict__ src,
                                    half_t* __restrict__ dst,
                                    int K, int N) {
  __shared__ float tile[32][33];
  int n0 = blockIdx.x * 32;
  int k0 = blockIdx.y * 32;
  int tx = threadIdx.x & 31;
  int ty = threadIdx.x >> 5;   // 0..7
#pragma unroll
  for (int i = ty; i < 32; i += 8)
    tile[i][tx] = src[(size_t)(k0 + i) * N + (n0 + tx)];
  __syncthreads();
#pragma unroll
  for (int i = ty; i < 32; i += 8)
    dst[(size_t)(n0 + i) * K + (k0 + tx)] = (half_t)tile[tx][i];
}

// ---------------------------------------------------------------------------
// Fragment loaders (ISA §7.12.2 layouts), LDS rows padded to 40 halves (80 B).
// ---------------------------------------------------------------------------
#define LDS_LD 40

// 16-bit A 16x32: lane 0-15 = row M, K 0..7 in v0..3, K 16..23 in v4..7;
// lanes 16-31 same rows, K 8..15 / 24..31.
__device__ __forceinline__ v16h frag_a(const half_t* t, int rowbase, int lane) {
  int r  = rowbase + (lane & 15);
  int kb = (lane & 16) ? 8 : 0;
  const half_t* p = t + r * LDS_LD + kb;
  v8h lo = *(const v8h*)p;
  v8h hi = *(const v8h*)(p + 16);
  v16h f;
#pragma unroll
  for (int i = 0; i < 8; ++i) { f[i] = lo[i]; f[i + 8] = hi[i]; }
  return f;
}

// 16-bit B 32x16 (stored transposed, N x K): lane 0-15 = col N, K 0..7 / 16..23;
// lanes 16-31 = K 8..15 / 24..31.
__device__ __forceinline__ v16h frag_b(const half_t* t, int rowbase, int lane) {
  int r  = rowbase + (lane & 15);
  int kb = (lane & 16) ? 8 : 0;
  const half_t* p = t + r * LDS_LD + kb;
  v8h lo = *(const v8h*)p;
  v8h hi = *(const v8h*)(p + 16);
  v16h f;
#pragma unroll
  for (int i = 0; i < 8; ++i) { f[i] = lo[i]; f[i + 8] = hi[i]; }
  return f;
}

// ---------------------------------------------------------------------------
// Gated GEMM: O = act(X @ Wa^T + ba, X @ Wg^T + bg), act = tanh(a)*sigmoid(g)
//   X : M x K (f16, row-major),  Wa/Wg : N x K (f16, row-major = transposed)
// Block tile 128(M) x 128(N), BK = 32, 256 threads = 8 waves in a 4(M)x2(N)
// grid; each wave: 2 M-strips x 4 N-tiles x 2 gates = 16 v_wmma per K-step,
// with each B fragment feeding 2 WMMAs (ds:wmma = 1.25:1).
// Double-buffered LDS fed by global_load_async_to_lds_b128 (ASYNCcnt).
// MODE 0: write f16 activation to outH (ld = ldo)
// MODE 1: left final  -> outF: branch==0 -> val, branch==2 -> 0, branch==1 skip
// MODE 2: right final -> outF: branch==1 -> val, else skip
// ---------------------------------------------------------------------------
template <int MODE>
__global__ __launch_bounds__(256)
void k_gated_gemm(const half_t* __restrict__ X,
                  const half_t* __restrict__ Wa,
                  const half_t* __restrict__ Wg,
                  const float* __restrict__ ba,
                  const float* __restrict__ bg,
                  int M, int N, int K,
                  half_t* __restrict__ outH,
                  float*  __restrict__ outF,
                  int ldo,
                  const int* __restrict__ branch) {
  __shared__ __align__(16) half_t sA[2][128 * LDS_LD];
  __shared__ __align__(16) half_t sB[2][128 * LDS_LD];
  __shared__ __align__(16) half_t sG[2][128 * LDS_LD];

  const int tid    = threadIdx.x;
  const int lane   = tid & 31;
  const int wid    = tid >> 5;          // 0..7
  const int wave_m = wid >> 1;          // 0..3 -> 32-row strip pair
  const int wave_n = wid & 1;           // 0..1 -> 64-col half
  const int m0     = blockIdx.y * 128;
  const int n0     = blockIdx.x * 128;

  // Copy assignment: each 128x32 tile = 512 16B-chunks -> 2 chunks/thread.
  const int r0 = tid >> 2;              // rows 0..63
  const int r1 = r0 + 64;               // rows 64..127
  const int c8 = (tid & 3) * 8;         // column offset in halves
  const half_t* gA0 = X  + (size_t)(m0 + r0) * K + c8;
  const half_t* gA1 = X  + (size_t)(m0 + r1) * K + c8;
  const half_t* gB0 = Wa + (size_t)(n0 + r0) * K + c8;
  const half_t* gB1 = Wa + (size_t)(n0 + r1) * K + c8;
  const half_t* gG0 = Wg + (size_t)(n0 + r0) * K + c8;
  const half_t* gG1 = Wg + (size_t)(n0 + r1) * K + c8;

  unsigned lA0[2], lA1[2], lB0[2], lB1[2], lG0[2], lG1[2];
#pragma unroll
  for (int b = 0; b < 2; ++b) {
    lA0[b] = lds_off_u32(&sA[b][r0 * LDS_LD + c8]);
    lA1[b] = lds_off_u32(&sA[b][r1 * LDS_LD + c8]);
    lB0[b] = lds_off_u32(&sB[b][r0 * LDS_LD + c8]);
    lB1[b] = lds_off_u32(&sB[b][r1 * LDS_LD + c8]);
    lG0[b] = lds_off_u32(&sG[b][r0 * LDS_LD + c8]);
    lG1[b] = lds_off_u32(&sG[b][r1 * LDS_LD + c8]);
  }

  v8f acc[2][4][2] = {};   // [m-strip][n-tile][gate]

  // Prologue: async-fill buffer 0 with K-tile 0.
  async_ld_b128(lA0[0], gA0); async_ld_b128(lA1[0], gA1);
  async_ld_b128(lB0[0], gB0); async_ld_b128(lB1[0], gB1);
  async_ld_b128(lG0[0], gG0); async_ld_b128(lG1[0], gG1);

  const int nsteps = K >> 5;            // K / 32
  for (int i = 0; i < nsteps; ++i) {
    const int cur = i & 1;
    wait_async0();                      // my tile-i transfers landed in LDS
    __syncthreads();                    // everyone's landed; prev compute done

    if (i + 1 < nsteps) {               // overlap tile i+1 loads with compute
      const int kk = (i + 1) << 5;
      async_ld_b128(lA0[cur ^ 1], gA0 + kk); async_ld_b128(lA1[cur ^ 1], gA1 + kk);
      async_ld_b128(lB0[cur ^ 1], gB0 + kk); async_ld_b128(lB1[cur ^ 1], gB1 + kk);
      async_ld_b128(lG0[cur ^ 1], gG0 + kk); async_ld_b128(lG1[cur ^ 1], gG1 + kk);
    }

    const half_t* cA = &sA[cur][0];
    const half_t* cB = &sB[cur][0];
    const half_t* cG = &sG[cur][0];

    v16h af0 = frag_a(cA, wave_m * 32,      lane);
    v16h af1 = frag_a(cA, wave_m * 32 + 16, lane);
#pragma unroll
    for (int n = 0; n < 4; ++n) {
      int rb = wave_n * 64 + n * 16;
      v16h bfa = frag_b(cB, rb, lane);
      v16h bfg = frag_b(cG, rb, lane);
      acc[0][n][0] = __builtin_amdgcn_wmma_f32_16x16x32_f16(
          false, af0, false, bfa, (short)0, acc[0][n][0], false, false);
      acc[1][n][0] = __builtin_amdgcn_wmma_f32_16x16x32_f16(
          false, af1, false, bfa, (short)0, acc[1][n][0], false, false);
      acc[0][n][1] = __builtin_amdgcn_wmma_f32_16x16x32_f16(
          false, af0, false, bfg, (short)0, acc[0][n][1], false, false);
      acc[1][n][1] = __builtin_amdgcn_wmma_f32_16x16x32_f16(
          false, af1, false, bfg, (short)0, acc[1][n][1], false, false);
    }
  }

  // Epilogue. C/D layout: elem r, lane l -> M = r + (l>=16 ? 8 : 0), N = l&15
  const int rowoff = (lane & 16) ? 8 : 0;
  const int coln   = lane & 15;
#pragma unroll
  for (int n = 0; n < 4; ++n) {
    int col = n0 + wave_n * 64 + n * 16 + coln;
    float bav = ba[col];
    float bgv = bg[col];
#pragma unroll
    for (int m = 0; m < 2; ++m) {
#pragma unroll
      for (int r = 0; r < 8; ++r) {
        int row = m0 + wave_m * 32 + m * 16 + r + rowoff;
        float a = acc[m][n][0][r] + bav;
        float g = acc[m][n][1][r] + bgv;
        float val = tanhf(a) * (1.0f / (1.0f + __expf(-g)));
        if (MODE == 0) {
          outH[(size_t)row * ldo + col] = (half_t)val;
        } else if (MODE == 1) {
          int b = branch[row];
          if (b != 1) outF[(size_t)row * ldo + col] = (b == 0) ? val : 0.0f;
        } else {
          if (branch[row] == 1) outF[(size_t)row * ldo + col] = val;
        }
      }
    }
  }
}

// ---------------------------------------------------------------------------
extern "C" void kernel_launch(void* const* d_in, const int* in_sizes, int n_in,
                              void* d_out, int out_size, void* d_ws, size_t ws_size,
                              hipStream_t stream) {
  const float* hid   = (const float*)d_in[0];
  const float* ctx   = (const float*)d_in[1];
  const float* lab   = (const float*)d_in[2];
  const float* left  = (const float*)d_in[3];
  const int*   brch  = (const int*)  d_in[4];
  const float* lw_h1 = (const float*)d_in[5];  const float* lb_h1 = (const float*)d_in[6];
  const float* lw_g1 = (const float*)d_in[7];  const float* lb_g1 = (const float*)d_in[8];
  const float* lw_h2 = (const float*)d_in[9];  const float* lb_h2 = (const float*)d_in[10];
  const float* lw_g2 = (const float*)d_in[11]; const float* lb_g2 = (const float*)d_in[12];
  const float* rw_h1 = (const float*)d_in[13]; const float* rb_h1 = (const float*)d_in[14];
  const float* rw_g1 = (const float*)d_in[15]; const float* rb_g1 = (const float*)d_in[16];
  const float* rw_h2 = (const float*)d_in[17]; const float* rb_h2 = (const float*)d_in[18];
  const float* rw_g2 = (const float*)d_in[19]; const float* rb_g2 = (const float*)d_in[20];
  float* out = (float*)d_out;

  // Workspace layout (f16 buffers)
  char* ws = (char*)d_ws;
  size_t off = 0;
  half_t* X1    = (half_t*)(ws + off); off += (size_t)B_ROWS * 3 * HDIM * 2;   // 50.3 MB
  half_t* hL    = (half_t*)(ws + off); off += (size_t)B_ROWS * HDIM * 2;       // 16.8 MB
  half_t* X2r   = (half_t*)(ws + off); off += (size_t)B_ROWS * 2 * HDIM * 2;   // 33.6 MB
  half_t* lwh1t = (half_t*)(ws + off); off += (size_t)3 * HDIM * HDIM * 2;
  half_t* lwg1t = (half_t*)(ws + off); off += (size_t)3 * HDIM * HDIM * 2;
  half_t* rwh1t = (half_t*)(ws + off); off += (size_t)3 * HDIM * HDIM * 2;
  half_t* rwg1t = (half_t*)(ws + off); off += (size_t)3 * HDIM * HDIM * 2;
  half_t* lwh2t = (half_t*)(ws + off); off += (size_t)HDIM * HDIM * 2;
  half_t* lwg2t = (half_t*)(ws + off); off += (size_t)HDIM * HDIM * 2;
  half_t* rwh2t = (half_t*)(ws + off); off += (size_t)2 * HDIM * HDIM * 2;
  half_t* rwg2t = (half_t*)(ws + off); off += (size_t)2 * HDIM * HDIM * 2;

  // 1) Pack activations to f16
  k_pack_x1<<<(B_ROWS * 3 * HDIM) / 256, 256, 0, stream>>>(hid, ctx, lab, X1);
  k_pack_left<<<(B_ROWS * HDIM) / 256, 256, 0, stream>>>(left, X2r);

  // 2) Transpose-convert weights: (K x N) f32 -> (N x K) f16
  dim3 tb(256);
  k_transpose_f32_f16<<<dim3(HDIM / 32, 3 * HDIM / 32), tb, 0, stream>>>(lw_h1, lwh1t, 3 * HDIM, HDIM);
  k_transpose_f32_f16<<<dim3(HDIM / 32, 3 * HDIM / 32), tb, 0, stream>>>(lw_g1, lwg1t, 3 * HDIM, HDIM);
  k_transpose_f32_f16<<<dim3(HDIM / 32, 3 * HDIM / 32), tb, 0, stream>>>(rw_h1, rwh1t, 3 * HDIM, HDIM);
  k_transpose_f32_f16<<<dim3(HDIM / 32, 3 * HDIM / 32), tb, 0, stream>>>(rw_g1, rwg1t, 3 * HDIM, HDIM);
  k_transpose_f32_f16<<<dim3(HDIM / 32, HDIM / 32),     tb, 0, stream>>>(lw_h2, lwh2t, HDIM, HDIM);
  k_transpose_f32_f16<<<dim3(HDIM / 32, HDIM / 32),     tb, 0, stream>>>(lw_g2, lwg2t, HDIM, HDIM);
  k_transpose_f32_f16<<<dim3(HDIM / 32, 2 * HDIM / 32), tb, 0, stream>>>(rw_h2, rwh2t, 2 * HDIM, HDIM);
  k_transpose_f32_f16<<<dim3(HDIM / 32, 2 * HDIM / 32), tb, 0, stream>>>(rw_g2, rwg2t, 2 * HDIM, HDIM);

  // 3) Stage-1 gated GEMMs (K = 3072): h_l, and h_r into X2r cols [0,1024)
  dim3 g1(HDIM / 128, B_ROWS / 128);
  k_gated_gemm<0><<<g1, 256, 0, stream>>>(X1, lwh1t, lwg1t, lb_h1, lb_g1,
                                          B_ROWS, HDIM, 3 * HDIM,
                                          hL, nullptr, HDIM, nullptr);
  k_gated_gemm<0><<<g1, 256, 0, stream>>>(X1, rwh1t, rwg1t, rb_h1, rb_g1,
                                          B_ROWS, HDIM, 3 * HDIM,
                                          X2r, nullptr, 2 * HDIM, nullptr);

  // 4) Stage-2 gated GEMMs with fused branch-select epilogue into d_out
  k_gated_gemm<1><<<g1, 256, 0, stream>>>(hL, lwh2t, lwg2t, lb_h2, lb_g2,
                                          B_ROWS, HDIM, HDIM,
                                          nullptr, out, HDIM, brch);
  k_gated_gemm<2><<<g1, 256, 0, stream>>>(X2r, rwh2t, rwg2t, rb_h2, rb_g2,
                                          B_ROWS, HDIM, 2 * HDIM,
                                          nullptr, out, HDIM, brch);
}
